// AlignmentAttentionModule_18124761989825
// MI455X (gfx1250) — matrix operand order repaired and checked by reading, not compiled, not run on hardware
//
#include <hip/hip_runtime.h>

// ---------------------------------------------------------------------------
// Problem constants (from the reference)
// ---------------------------------------------------------------------------
constexpr int S    = 512;   // seq len (lm == am)
constexpr int NB   = 64;    // batch
constexpr int NH   = 5;     // heads
constexpr int QD   = 32;    // q/k head dim
constexpr int PD   = 4;     // pos head dim
constexpr int VD   = 12;    // value head dim
constexpr int M_SB = S * NB;            // 32768 rows for row-major GEMMs
constexpr int LM_N    = (QD + PD) * NH; // 180
constexpr int LM_NPAD = 192;
constexpr int AM_N    = QD * NH;        // 160
constexpr int V_N     = VD * NH;        // 60
constexpr int V_STRIDE = 80;            // per-head padded: col = h*16 + d
constexpr int PE_ROWS = 2 * S - 1;      // 1023

typedef __bf16 bf16;
typedef __attribute__((ext_vector_type(16))) __bf16 v16bf;
typedef __attribute__((ext_vector_type(8)))  float  v8f;

// A-matrix (16x32 bf16) lane/element -> K mapping (ISA 7.12.2, 16-bit A 16x32)
__device__ __forceinline__ int a_k(int e, int half) {
  int v = e >> 1, pos = e & 1;
  return (v < 4) ? (8 * half + 2 * v + pos) : (16 + 8 * half + 2 * (v - 4) + pos);
}
// B-matrix (32x16 bf16): same K mapping as A.
__device__ __forceinline__ int b_k(int e, int half) { return a_k(e, half); }

// ---- CDNA5 async global->LDS copy (ASYNCcnt path), ISA 15.18.3 op 98 ------
// LDS destination is the byte offset held in the vdst VGPR; source is a 64-bit global address
__device__ __forceinline__ void async_copy_b128(unsigned lds_byte_off, const void* gaddr) {
  asm volatile("global_load_async_to_lds_b128 %0, %1, off"
               :: "v"(lds_byte_off), "v"(gaddr)
               : "memory");
}
__device__ __forceinline__ void wait_async0() {
  asm volatile("s_wait_asynccnt 0" ::: "memory");
}

// ---------------------------------------------------------------------------
// pe = pos_emb @ W_pos   (1023 x 192) x (192 x 20)  -- tiny, scalar kernel
// ---------------------------------------------------------------------------
__global__ __launch_bounds__(256) void pe_kernel(const float* __restrict__ pos_emb,
                                                 const float* __restrict__ W_pos,
                                                 float* __restrict__ pe) {
  int idx = blockIdx.x * blockDim.x + threadIdx.x;
  if (idx >= PE_ROWS * 20) return;
  int n = idx / 20, c = idx % 20;
  float acc = 0.f;
  const float* pr = pos_emb + (size_t)n * 192;
  for (int i = 0; i < 192; ++i) acc += pr[i] * W_pos[i * 20 + c];
  pe[idx] = acc;
}

// ---------------------------------------------------------------------------
// Projection GEMM: C(bf16, M x Npad) = A(f32, M x K) @ W(f32, K x N) + b
// block = 256 threads = 8 waves; each wave owns a 16x16 tile; block = 128 x 16.
// A tile staged via async global->LDS b128 (no VGPR round trip).
// vmap != 0 : remap output col c -> (c/12)*16 + c%12  (value projection)
// ---------------------------------------------------------------------------
__global__ __launch_bounds__(256)
void proj_gemm(const float* __restrict__ A, const float* __restrict__ W,
               const float* __restrict__ bias, bf16* __restrict__ C,
               int M, int K, int N, int Npad, int vmap) {
  __shared__ char sbuf[128 * 32 * 4 + 32 * 16 * 4];
  float* At = (float*)sbuf;                    // 128 x 32, LDS byte off 0
  float* Wt = (float*)(sbuf + 128 * 32 * 4);   // 32 x 16
  constexpr unsigned WT_OFF = 128 * 32 * 4;
  (void)WT_OFF;
  const int tid = threadIdx.x, lane = tid & 31, wave = tid >> 5;
  const int m0 = blockIdx.x * 128;
  const int n0 = blockIdx.y * 16;
  const int nl = lane & 15, half = lane >> 4;
  v8f acc = {0.f, 0.f, 0.f, 0.f, 0.f, 0.f, 0.f, 0.f};

  for (int kc = 0; kc < K; kc += 32) {
    if (kc + 32 < K) __builtin_prefetch(&A[(size_t)(m0 + (tid >> 1)) * K + kc + 32], 0, 1);
    // A tile: 128 rows x 32 f32 = 16KB, 8 x 16B chunks per row, 4 per thread
    for (int j = 0; j < 4; ++j) {
      int i = tid + 256 * j;
      int r = i >> 3, chunk = i & 7;
      const char* g = (const char*)A + ((size_t)(m0 + r) * K + kc) * 4 + chunk * 16;
      async_copy_b128((unsigned)(i * 16), g);
    }
    // W tile (needs N-bounds zero fill): scalar
    for (int j = 0; j < 2; ++j) {
      int e = tid + 256 * j;
      int r = e >> 4, c = e & 15;
      Wt[e] = (n0 + c < N) ? W[(size_t)(kc + r) * N + n0 + c] : 0.f;
    }
    wait_async0();
    __syncthreads();
    v16bf af, bfr;
    for (int e = 0; e < 16; ++e) af[e]  = (bf16)At[(wave * 16 + nl) * 32 + a_k(e, half)];
    for (int e = 0; e < 16; ++e) bfr[e] = (bf16)Wt[b_k(e, half) * 16 + nl];
    acc = __builtin_amdgcn_wmma_f32_16x16x32_bf16(false, af, false, bfr, (short)0, acc,
                                                  false, false);
    __syncthreads();
  }
  int c0 = n0 + nl;
  if (c0 < N) {
    float bv = bias[c0];
    int dstc = vmap ? (c0 / VD) * 16 + (c0 % VD) : c0;
    for (int r = 0; r < 8; ++r) {
      int m = r + 8 * half;
      C[(size_t)(m0 + wave * 16 + m) * Npad + dstc] = (bf16)(acc[r] + bv);
    }
  }
}

// ---------------------------------------------------------------------------
// Fused rel-position attention for one (h, b, 64-row t-block).
// Q/K/V staged with async global->LDS b128; K (32KB), V (16KB), the 64x512
// f32 score block (128KB) and 64x512 bf16 weights (64KB) stay in LDS.
// ---------------------------------------------------------------------------
__global__ __launch_bounds__(256)
void attn_kernel(const bf16* __restrict__ lmp,   // (S*NB) x 192  (q | p)
                 const bf16* __restrict__ amp,   // (S*NB) x 160  (k)
                 const bf16* __restrict__ vp,    // (S*NB) x 80   (v, per-head 16-pad)
                 const float* __restrict__ pe,   // 1023 x 20
                 bf16* __restrict__ attnout) {   // (S*NB) x 64
  const int t0 = blockIdx.x * 64;
  const int b  = blockIdx.y;
  const int h  = blockIdx.z;
  const int tid = threadIdx.x, lane = tid & 31, wave = tid >> 5;
  const int nl = lane & 15, half = lane >> 4;

  extern __shared__ char smem[];
  constexpr unsigned QS_OFF = 0;                         // 64 x 32 bf16
  constexpr unsigned KS_OFF = QS_OFF + 64 * 32 * 2;      // 512 x 32 bf16
  constexpr unsigned VS_OFF = KS_OFF + 512 * 32 * 2;     // 512 x 16 bf16
  constexpr unsigned PS_OFF = VS_OFF + 512 * 16 * 2;     // 64 x 4 f32
  constexpr unsigned SC_OFF = PS_OFF + 64 * 4 * 4;       // 64 x 512 f32
  constexpr unsigned WB_OFF = SC_OFF + 64 * 512 * 4;     // 64 x 512 bf16
  bf16*  Qs = (bf16*)(smem + QS_OFF);
  bf16*  Ks = (bf16*)(smem + KS_OFF);
  bf16*  Vs = (bf16*)(smem + VS_OFF);
  float* Ps = (float*)(smem + PS_OFF);
  float* Sc = (float*)(smem + SC_OFF);
  bf16*  Wb = (bf16*)(smem + WB_OFF);

  // ---- async stage Q (256 x 16B), K (2048 x 16B), V (1024 x 16B) -------
  {
    int i = tid;                         // Q: row = i>>2, 16B chunk = i&3
    int t = i >> 2, chunk = i & 3;
    const char* g = (const char*)lmp +
        ((size_t)((t0 + t) * NB + b) * LM_NPAD + h * QD) * 2 + chunk * 16;
    async_copy_b128(QS_OFF + (unsigned)(i * 16), g);
  }
  for (int j = 0; j < 8; ++j) {          // K: row = i>>2, chunk = i&3
    int i = tid + 256 * j;
    int s = i >> 2, chunk = i & 3;
    const char* g = (const char*)amp +
        ((size_t)(s * NB + b) * AM_N + h * QD) * 2 + chunk * 16;
    async_copy_b128(KS_OFF + (unsigned)(i * 16), g);
  }
  for (int j = 0; j < 4; ++j) {          // V: row = i>>1, chunk = i&1
    int i = tid + 256 * j;
    int s = i >> 1, chunk = i & 1;
    const char* g = (const char*)vp +
        ((size_t)(s * NB + b) * V_STRIDE + h * 16) * 2 + chunk * 16;
    async_copy_b128(VS_OFF + (unsigned)(i * 16), g);
  }
  {                                      // P: 64 x 4 (scalar, 8B rows)
    int t = tid >> 2, j = tid & 3;
    Ps[tid] = (float)lmp[(size_t)((t0 + t) * NB + b) * LM_NPAD + QD * NH + h * PD + j];
  }
  wait_async0();
  __syncthreads();

  // ---- content scores: Q(64x32) x K^T(32x512), one WMMA per 16x16 tile ----
  const int tt   = wave & 3;   // this wave's fixed t-tile (0..3)
  const int sgrp = wave >> 2;  // s-tile group (0..1), 16 tiles each
  v16bf aq;
  for (int e = 0; e < 16; ++e) aq[e] = Qs[(tt * 16 + nl) * 32 + a_k(e, half)];
  for (int j = 0; j < 16; ++j) {
    int ss = sgrp * 16 + j;
    v16bf bk;
    for (int e = 0; e < 16; ++e) bk[e] = Ks[(ss * 16 + nl) * 32 + b_k(e, half)];
    v8f c = {0.f, 0.f, 0.f, 0.f, 0.f, 0.f, 0.f, 0.f};
    c = __builtin_amdgcn_wmma_f32_16x16x32_bf16(false, aq, false, bk, (short)0, c,
                                                false, false);
    for (int r = 0; r < 8; ++r) {
      int m = r + 8 * half;
      Sc[(tt * 16 + m) * 512 + ss * 16 + nl] = c[r];
    }
  }
  __syncthreads();

  // ---- rel-shift positional term: Sc[t][s] += p[t] . pe[S-1-t0-t+s] ----
  for (int i = 0; i < 128; ++i) {
    int e = tid + 256 * i;
    int t = e >> 9, s = e & 511;
    int n = (S - 1) - (t0 + t) + s;      // in [0, 1022]
    const float* per = pe + (size_t)n * 20 + h * PD;
    float acc = Ps[t * 4 + 0] * per[0] + Ps[t * 4 + 1] * per[1] +
                Ps[t * 4 + 2] * per[2] + Ps[t * 4 + 3] * per[3];
    Sc[e] += acc;
  }
  __syncthreads();

  // ---- row softmax over 512 cols; emit bf16 weights -------------------
  for (int rr = 0; rr < 8; ++rr) {
    int row = wave * 8 + rr;
    float vals[16];
    float mx = -1e30f;
    for (int i = 0; i < 16; ++i) {
      vals[i] = Sc[row * 512 + lane + 32 * i];
      mx = fmaxf(mx, vals[i]);
    }
    for (int o = 16; o > 0; o >>= 1) mx = fmaxf(mx, __shfl_xor(mx, o, 32));
    float sum = 0.f;
    for (int i = 0; i < 16; ++i) { vals[i] = __expf(vals[i] - mx); sum += vals[i]; }
    for (int o = 16; o > 0; o >>= 1) sum += __shfl_xor(sum, o, 32);
    float inv = 1.f / sum;
    for (int i = 0; i < 16; ++i)
      Wb[row * 512 + lane + 32 * i] = (bf16)(vals[i] * inv);
  }
  __syncthreads();

  // ---- out = W(64x512) x V(512x16); waves 0..3, K-loop of 16 WMMAs -----
  if (wave < 4) {
    v8f acc = {0.f, 0.f, 0.f, 0.f, 0.f, 0.f, 0.f, 0.f};
    for (int kc = 0; kc < 16; ++kc) {
      v16bf aw, bv;
      for (int e = 0; e < 16; ++e)
        aw[e] = Wb[(wave * 16 + nl) * 512 + kc * 32 + a_k(e, half)];
      for (int e = 0; e < 16; ++e)
        bv[e] = Vs[(kc * 32 + b_k(e, half)) * 16 + nl];
      acc = __builtin_amdgcn_wmma_f32_16x16x32_bf16(false, aw, false, bv, (short)0, acc,
                                                    false, false);
    }
    if (nl < VD) {
      for (int r = 0; r < 8; ++r) {
        int m = r + 8 * half;
        int t = t0 + wave * 16 + m;
        attnout[(size_t)(t * NB + b) * 64 + h * VD + nl] = (bf16)acc[r];
      }
    }
  }
}

// ---------------------------------------------------------------------------
// Final projection: out(f32, 32768x512) = attnout(bf16, 32768x64, K<60 valid)
//                                          @ W_out(f32, 60x512) + b_out
// W rows 60..63 zeroed so junk pad cols of attnout contribute nothing.
// ---------------------------------------------------------------------------
__global__ __launch_bounds__(256)
void out_gemm(const bf16* __restrict__ A, const float* __restrict__ W,
              const float* __restrict__ bias, float* __restrict__ C) {
  __shared__ char sbuf[128 * 64 * 2 + 64 * 16 * 4];
  bf16*  At = (bf16*)sbuf;                     // 128 x 64, LDS byte off 0
  float* Wt = (float*)(sbuf + 128 * 64 * 2);   // 64 x 16
  const int tid = threadIdx.x, lane = tid & 31, wave = tid >> 5;
  const int m0 = blockIdx.x * 128;
  const int n0 = blockIdx.y * 16;
  const int nl = lane & 15, half = lane >> 4;

  // A tile: 128 rows x 128B = 8 x 16B chunks per row, async
  for (int j = 0; j < 4; ++j) {
    int i = tid + 256 * j;
    int r = i >> 3, chunk = i & 7;
    const char* g = (const char*)A + (size_t)(m0 + r) * 128 + chunk * 16;
    async_copy_b128((unsigned)(i * 16), g);
  }
  for (int j = 0; j < 4; ++j) {
    int e = tid + 256 * j;
    int r = e >> 4, c = e & 15;
    Wt[e] = (r < V_N) ? W[(size_t)r * 512 + n0 + c] : 0.f;
  }
  wait_async0();
  __syncthreads();

  v8f acc = {0.f, 0.f, 0.f, 0.f, 0.f, 0.f, 0.f, 0.f};
  for (int kc = 0; kc < 64; kc += 32) {
    v16bf af, bfr;
    for (int e = 0; e < 16; ++e) af[e]  = At[(wave * 16 + nl) * 64 + kc + a_k(e, half)];
    for (int e = 0; e < 16; ++e) bfr[e] = (bf16)Wt[(kc + b_k(e, half)) * 16 + nl];
    acc = __builtin_amdgcn_wmma_f32_16x16x32_bf16(false, af, false, bfr, (short)0, acc,
                                                  false, false);
  }
  float bv = bias[n0 + nl];
  for (int r = 0; r < 8; ++r) {
    int m = r + 8 * half;
    C[(size_t)(m0 + wave * 16 + m) * 512 + n0 + nl] = acc[r] + bv;
  }
}

// ---------------------------------------------------------------------------
// Host launcher
// ---------------------------------------------------------------------------
extern "C" void kernel_launch(void* const* d_in, const int* in_sizes, int n_in,
                              void* d_out, int out_size, void* d_ws, size_t ws_size,
                              hipStream_t stream) {
  const float* lm_pruned = (const float*)d_in[0];
  const float* am_pruned = (const float*)d_in[1];
  const float* pos_emb   = (const float*)d_in[2];
  const float* W_lm      = (const float*)d_in[3];
  const float* b_lm      = (const float*)d_in[4];
  const float* W_am      = (const float*)d_in[5];
  const float* b_am      = (const float*)d_in[6];
  const float* W_pos     = (const float*)d_in[7];
  const float* W_in      = (const float*)d_in[8];
  const float* b_in      = (const float*)d_in[9];
  const float* W_out     = (const float*)d_in[10];
  const float* b_out     = (const float*)d_in[11];
  float* out = (float*)d_out;

  // workspace carve
  char* ws = (char*)d_ws;
  bf16* lm_proj = (bf16*)ws;                                           // 32768*192*2
  bf16* am_proj = (bf16*)(ws + (size_t)M_SB * LM_NPAD * 2);            // 32768*160*2
  bf16* v_proj  = (bf16*)(ws + (size_t)M_SB * (LM_NPAD + AM_N) * 2);   // 32768*80*2
  float* pe     = (float*)(ws + (size_t)M_SB * (LM_NPAD + AM_N + V_STRIDE) * 2);
  bf16* attnout = (bf16*)((char*)pe + 81920);                          // 32768*64*2

  // 1) positional embedding projection (tiny)
  pe_kernel<<<dim3((PE_ROWS * 20 + 255) / 256), dim3(256), 0, stream>>>(pos_emb, W_pos, pe);

  // 2) three input projections (bf16 WMMA, f32 accumulate)
  proj_gemm<<<dim3(M_SB / 128, (LM_N + 15) / 16), dim3(256), 0, stream>>>(
      lm_pruned, W_lm, b_lm, lm_proj, M_SB, 512, LM_N, LM_NPAD, 0);
  proj_gemm<<<dim3(M_SB / 128, (AM_N + 15) / 16), dim3(256), 0, stream>>>(
      am_pruned, W_am, b_am, am_proj, M_SB, 512, AM_N, AM_N, 0);
  proj_gemm<<<dim3(M_SB / 128, (V_N + 15) / 16), dim3(256), 0, stream>>>(
      am_pruned, W_in, b_in, v_proj, M_SB, 512, V_N, V_STRIDE, 1);

  // 3) fused attention: scores + rel-shift + softmax + @V, all in LDS
  constexpr size_t ATTN_LDS =
      (64 * 32 + 512 * 32 + 512 * 16) * sizeof(bf16) +  // Q,K,V
      64 * 4 * sizeof(float) +                          // P
      64 * 512 * sizeof(float) +                        // scores
      64 * 512 * sizeof(bf16);                          // bf16 weights  => 250880
  (void)hipFuncSetAttribute((const void*)attn_kernel,
                            hipFuncAttributeMaxDynamicSharedMemorySize, (int)ATTN_LDS);
  attn_kernel<<<dim3(S / 64, NB, NH), dim3(256), ATTN_LDS, stream>>>(
      lm_proj, am_proj, v_proj, pe, attnout);

  // 4) output projection to f32
  out_gemm<<<dim3(M_SB / 128, 512 / 16), dim3(256), 0, stream>>>(attnout, W_out, b_out, out);
}
